// ContrastLossLocal_11605001634274
// MI455X (gfx1250) — matrix operand order repaired and checked by reading, not compiled, not run on hardware
//
#include <hip/hip_runtime.h>
#include <hip/hip_bf16.h>
#include <math.h>

// Problem constants (from reference)
#define C_ 19
#define U_ 32
#define N_ 2048
#define D_ 256
#define A_ 4096
#define P_ (C_ * U_)   // 608, divisible by 16

constexpr float kInvT = 10.0f;      // 1 / TEMPERATURE
constexpr float kEPS  = 1e-8f;
#define NEG_BIG (-1.0e30f)

typedef __attribute__((ext_vector_type(16))) __bf16 v16bf;
typedef __attribute__((ext_vector_type(8)))  __bf16 v8bf;
typedef __attribute__((ext_vector_type(4)))  __bf16 v4bf;
typedef __attribute__((ext_vector_type(8)))  float  v8f;
typedef int v4i __attribute__((vector_size(16)));   // builtin's b128 payload type

#define AS1 __attribute__((address_space(1)))
#define AS3 __attribute__((address_space(3)))

// gfx1250 async global->LDS path (ASYNCcnt), with sync fallback
#if __has_builtin(__builtin_amdgcn_global_load_async_to_lds_b128) && \
    __has_builtin(__builtin_amdgcn_s_wait_asynccnt)
#define USE_ASYNC 1
#define WAIT_ASYNC(n) __builtin_amdgcn_s_wait_asynccnt(n)
#else
#define USE_ASYNC 0
#define WAIT_ASYNC(n)
#endif

// ---------------------------------------------------------------------------
// WMMA helpers
// ---------------------------------------------------------------------------
__device__ inline v8f wmma_bf(const v16bf& a, const v16bf& b, v8f acc) {
  return __builtin_amdgcn_wmma_f32_16x16x32_bf16(false, a, false, b, (short)0, acc,
                                                 false, false);
}

// A-matrix 16x32 bf16 fragment (wave32 layout, ISA 7.12.2):
//   lane&15 = M row; lanes>=16 shift K by +8; e<8 -> K=k0+e, e>=8 -> K=k0+16+(e-8)
__device__ inline v16bf load_fragA_bf(const __bf16* __restrict__ row, int kb, int lane) {
  const int k0 = kb + ((lane & 16) >> 1);   // +8 for upper half-wave
  v8bf a = *reinterpret_cast<const v8bf*>(row + k0);        // 16B aligned
  v8bf b = *reinterpret_cast<const v8bf*>(row + k0 + 16);
  return __builtin_shufflevector(a, b, 0, 1, 2, 3, 4, 5, 6, 7,
                                       8, 9, 10, 11, 12, 13, 14, 15);
}

// B-matrix 32x16 bf16 fragment from an LDS panel [16 rows x 256 K]:
// lane&15 = panel row (N col); lanes>=16 shift K by +8; e<8 -> K=k0+e, e>=8 -> K=k0+16+(e-8)
__device__ inline v16bf load_fragB_lds(const __bf16* panel, int kb, int lane) {
  const int k0 = kb + ((lane & 16) >> 1);
  const __bf16* p = panel + (lane & 15) * D_ + k0;
  v8bf a = *reinterpret_cast<const v8bf*>(p);        // 16B aligned
  v8bf b = *reinterpret_cast<const v8bf*>(p + 16);
  return __builtin_shufflevector(a, b, 0, 1, 2, 3, 4, 5, 6, 7,
                                       8, 9, 10, 11, 12, 13, 14, 15);
}

// Stage one 8 KB contiguous panel (16 rows x 256 bf16) into LDS with 256 threads:
// 2 chunks of 16B per thread. Async (ASYNCcnt) when available.
__device__ inline void stage8k(const __bf16* __restrict__ g, __bf16* s, int t) {
#pragma unroll
  for (int ch = 0; ch < 2; ++ch) {
    const int o = ch * 4096 + t * 16;
#if USE_ASYNC
    __builtin_amdgcn_global_load_async_to_lds_b128(
        (AS1 v4i*)(void*)((const char*)g + o),
        (AS3 v4i*)((char*)s + o), 0, 0);
#else
    *reinterpret_cast<float4*>((char*)s + o) =
        *reinterpret_cast<const float4*>((const char*)g + o);
#endif
  }
}

// ---------------------------------------------------------------------------
// KC: one-pass f32 -> bf16 hi/lo split (error ~2^-16 relative per product)
// ---------------------------------------------------------------------------
__global__ __launch_bounds__(256) void kc_split(const float* __restrict__ src,
                                                __bf16* __restrict__ hi,
                                                __bf16* __restrict__ lo, int n) {
  const int stride = gridDim.x * blockDim.x;
  for (int i = blockIdx.x * blockDim.x + threadIdx.x; i * 4 < n; i += stride) {
    float4 x = *reinterpret_cast<const float4*>(src + (size_t)i * 4);
    v4bf h, l;
    h[0] = (__bf16)x.x; l[0] = (__bf16)(x.x - (float)h[0]);
    h[1] = (__bf16)x.y; l[1] = (__bf16)(x.y - (float)h[1]);
    h[2] = (__bf16)x.z; l[2] = (__bf16)(x.z - (float)h[2]);
    h[3] = (__bf16)x.w; l[3] = (__bf16)(x.w - (float)h[3]);
    *reinterpret_cast<v4bf*>(hi + (size_t)i * 4) = h;
    *reinterpret_cast<v4bf*>(lo + (size_t)i * 4) = l;
  }
}

// ---------------------------------------------------------------------------
// K0: group anchors by class (deterministic: per-class sequential scan)
// ---------------------------------------------------------------------------
__global__ void k0_index(const int* __restrict__ labels,
                         int* __restrict__ order, int* __restrict__ cls_off) {
  __shared__ int cnts[C_];
  const int t = threadIdx.x;
  if (t < C_) {
    int cn = 0;
    for (int i = 0; i < A_; ++i) cn += (labels[i] == t);
    cnts[t] = cn;
  }
  __syncthreads();
  if (t == 0) {
    int s = 0;
    for (int c = 0; c < C_; ++c) { cls_off[c] = s; s += cnts[c]; }
    cls_off[C_] = s;
  }
  __syncthreads();
  if (t < C_) {
    int w = cls_off[t];
    for (int i = 0; i < A_; ++i)
      if (labels[i] == t) order[w++] = i;
  }
}

// ---------------------------------------------------------------------------
// K1: dots_p row max over valid prototypes (full A x P GEMM, split-bf16 WMMA).
// 8 waves/block share one double-buffered LDS B panel (async staged); each
// wave owns a 16-anchor M tile with A fragments pre-staged in VGPRs.
// ---------------------------------------------------------------------------
__global__ __launch_bounds__(256) void k1_protomax(const __bf16* __restrict__ a_hi,
                                                   const __bf16* __restrict__ a_lo,
                                                   const __bf16* __restrict__ p_hi,
                                                   const __bf16* __restrict__ p_lo,
                                                   const int* __restrict__ pmask,
                                                   float* __restrict__ lmax) {
  __shared__ __align__(16) __bf16 sB[2][2][16 * D_];  // [buf][hi/lo][16x256] = 32 KB
  const int t     = threadIdx.x;
  const int lane  = t & 31;
  const int wave  = t >> 5;
  const int mbase = (blockIdx.x * 8 + wave) * 16;
  const size_t arow = (size_t)(mbase + (lane & 15)) * D_;

  v16bf Ah[8], Al[8];
#pragma unroll
  for (int k = 0; k < 8; ++k) {
    Ah[k] = load_fragA_bf(a_hi + arow, k * 32, lane);
    Al[k] = load_fragA_bf(a_lo + arow, k * 32, lane);
  }

  float vmax[8];
#pragma unroll
  for (int r = 0; r < 8; ++r) vmax[r] = NEG_BIG;

  constexpr int NT = P_ / 16;  // 38
  // Prologue: stage panels 0 and 1
  stage8k(p_hi, &sB[0][0][0], t);
  stage8k(p_lo, &sB[0][1][0], t);
  stage8k(p_hi + (size_t)16 * D_, &sB[1][0][0], t);
  stage8k(p_lo + (size_t)16 * D_, &sB[1][1][0], t);

  int cur = 0;
  for (int i = 0; i < NT; ++i) {
    if (i + 1 < NT) { WAIT_ASYNC(4); } else { WAIT_ASYNC(0); }
    __syncthreads();

    const int ncol = i * 16 + (lane & 15);
    v8f acc = {};
#pragma unroll
    for (int k = 0; k < 8; ++k) {
      v16bf bh = load_fragB_lds(&sB[cur][0][0], k * 32, lane);
      v16bf bl = load_fragB_lds(&sB[cur][1][0], k * 32, lane);
      acc = wmma_bf(Ah[k], bh, acc);
      acc = wmma_bf(Ah[k], bl, acc);
      acc = wmma_bf(Al[k], bh, acc);
    }
    const float colpen = pmask[ncol] ? 0.0f : NEG_BIG;  // invalid column -> -inf
#pragma unroll
    for (int r = 0; r < 8; ++r)
      vmax[r] = fmaxf(vmax[r], acc[r] * kInvT + colpen);

    __syncthreads();
    if (i + 2 < NT) {
      stage8k(p_hi + (size_t)(i + 2) * 16 * D_, &sB[cur][0][0], t);
      stage8k(p_lo + (size_t)(i + 2) * 16 * D_, &sB[cur][1][0], t);
    }
    cur ^= 1;
  }
  // max over the 16 lanes that share each output row (halves are independent)
#pragma unroll
  for (int o = 1; o < 16; o <<= 1) {
#pragma unroll
    for (int r = 0; r < 8; ++r)
      vmax[r] = fmaxf(vmax[r], __shfl_xor(vmax[r], o, 32));
  }
  if ((lane & 15) == 0) {
    const int rowoff = mbase + ((lane >> 4) << 3);  // lanes0-15: rows 0-7, lane16: 8-15
#pragma unroll
    for (int r = 0; r < 8; ++r) lmax[rowoff + r] = vmax[r];
  }
}

// ---------------------------------------------------------------------------
// K2: per-class negative sums. GEMM (gathered same-class anchors) x localsT,
// accumulating exp(dot/T - rowmax). 8 waves/block share the async-staged
// LDS B panel; out-of-range waves compute harmlessly and mask their stores.
// ---------------------------------------------------------------------------
__global__ __launch_bounds__(256) void k2_neg(const __bf16* __restrict__ a_hi,
                                              const __bf16* __restrict__ a_lo,
                                              const __bf16* __restrict__ l_hi,
                                              const __bf16* __restrict__ l_lo,
                                              const int* __restrict__ order,
                                              const int* __restrict__ cls_off,
                                              const float* __restrict__ lmax,
                                              float* __restrict__ negs) {
  __shared__ __align__(16) __bf16 sB[2][2][16 * D_];  // 32 KB
  const int c    = blockIdx.x;
  const int t    = threadIdx.x;
  const int lane = t & 31;
  const int wave = t >> 5;
  const int off  = cls_off[c];
  const int cnt  = cls_off[c + 1] - off;
  if (blockIdx.y * 128 >= cnt) return;   // block-uniform early exit (before barriers)

  const int mtile = (blockIdx.y * 8 + wave) * 16;
  const bool wactive = (mtile < cnt);

  int mA = mtile + (lane & 15);
  if (mA > cnt - 1) mA = cnt - 1;        // clamp (also covers inactive waves)
  const size_t arow = (size_t)order[off + mA] * D_;

  v16bf Ah[8], Al[8];
#pragma unroll
  for (int k = 0; k < 8; ++k) {
    Ah[k] = load_fragA_bf(a_hi + arow, k * 32, lane);
    Al[k] = load_fragA_bf(a_lo + arow, k * 32, lane);
  }

  int aidx[8]; float amax[8]; bool rv[8]; float s[8];
  const int rbase = mtile + ((lane >> 4) << 3);
#pragma unroll
  for (int r = 0; r < 8; ++r) {
    const int m = rbase + r;
    rv[r]   = wactive && (m < cnt);
    aidx[r] = order[off + (rv[r] ? m : 0)];
    amax[r] = lmax[aidx[r]];
    s[r]    = 0.0f;
  }

  const __bf16* srcH = l_hi + (size_t)c * (N_ * D_);
  const __bf16* srcL = l_lo + (size_t)c * (N_ * D_);
  constexpr int NT = N_ / 16;  // 128

  stage8k(srcH, &sB[0][0][0], t);
  stage8k(srcL, &sB[0][1][0], t);
  stage8k(srcH + (size_t)16 * D_, &sB[1][0][0], t);
  stage8k(srcL + (size_t)16 * D_, &sB[1][1][0], t);

  int cur = 0;
  for (int i = 0; i < NT; ++i) {
    if (i + 1 < NT) { WAIT_ASYNC(4); } else { WAIT_ASYNC(0); }
    __syncthreads();

    v8f acc = {};
#pragma unroll
    for (int k = 0; k < 8; ++k) {
      v16bf bh = load_fragB_lds(&sB[cur][0][0], k * 32, lane);
      v16bf bl = load_fragB_lds(&sB[cur][1][0], k * 32, lane);
      acc = wmma_bf(Ah[k], bh, acc);
      acc = wmma_bf(Ah[k], bl, acc);
      acc = wmma_bf(Al[k], bh, acc);
    }
#pragma unroll
    for (int r = 0; r < 8; ++r) s[r] += __expf(acc[r] * kInvT - amax[r]);

    __syncthreads();
    if (i + 2 < NT) {
      stage8k(srcH + (size_t)(i + 2) * 16 * D_, &sB[cur][0][0], t);
      stage8k(srcL + (size_t)(i + 2) * 16 * D_, &sB[cur][1][0], t);
    }
    cur ^= 1;
  }
  // sum over the 16 lanes holding each row's columns
#pragma unroll
  for (int o = 1; o < 16; o <<= 1) {
#pragma unroll
    for (int r = 0; r < 8; ++r) s[r] += __shfl_xor(s[r], o, 32);
  }
  if ((lane & 15) == 0) {
#pragma unroll
    for (int r = 0; r < 8; ++r)
      if (rv[r]) negs[aidx[r]] = s[r];
  }
}

// ---------------------------------------------------------------------------
// K3: positives epilogue. One wave per anchor; lane u = prototype u of the
// anchor's class (U == warpSize == 32). Full-f32 VALU dot (tiny work).
// ---------------------------------------------------------------------------
__global__ __launch_bounds__(256) void k3_pos(const float* __restrict__ anchors,
                                              const float* __restrict__ proto,
                                              const int* __restrict__ pmask,
                                              const int* __restrict__ labels,
                                              const float* __restrict__ lmax,
                                              const float* __restrict__ negs,
                                              float* __restrict__ mlpp) {
  const int lane = threadIdx.x & 31;
  const int a    = blockIdx.x * 8 + (threadIdx.x >> 5);
  const int c    = labels[a];
  const float mx = lmax[a];
  const float ng = negs[a];
  const int   p  = c * U_ + lane;
  const float* pp = proto + (size_t)p * D_;
  const float* ap = anchors + (size_t)a * D_;

  float dot = 0.0f;
  for (int d = 0; d < D_; d += 4) {
    float4 va = *reinterpret_cast<const float4*>(ap + d);
    float4 vb = *reinterpret_cast<const float4*>(pp + d);
    dot += va.x * vb.x + va.y * vb.y + va.z * vb.z + va.w * vb.w;
  }
  const float logit = dot * kInvT - mx;
  const bool  v = (pmask[p] != 0);
  float tt = v ? (logit - __logf(__expf(logit) + ng + kEPS)) : 0.0f;
  float cf = v ? 1.0f : 0.0f;
#pragma unroll
  for (int o = 16; o >= 1; o >>= 1) {
    tt += __shfl_xor(tt, o, 32);
    cf += __shfl_xor(cf, o, 32);
  }
  if (lane == 0) mlpp[a] = tt / (cf + kEPS);
}

// ---------------------------------------------------------------------------
// K4: deterministic fixed-order reduction to the scalar loss
// ---------------------------------------------------------------------------
__global__ __launch_bounds__(256) void k4_reduce(const float* __restrict__ mlpp,
                                                 float* __restrict__ out) {
  __shared__ float sm[256];
  float s = 0.0f;
  for (int i = threadIdx.x; i < A_; i += 256) s += mlpp[i];
  sm[threadIdx.x] = s;
  __syncthreads();
  for (int st = 128; st > 0; st >>= 1) {
    if (threadIdx.x < st) sm[threadIdx.x] += sm[threadIdx.x + st];
    __syncthreads();
  }
  if (threadIdx.x == 0) out[0] = -sm[0] / (float)A_;
}

// ---------------------------------------------------------------------------
extern "C" void kernel_launch(void* const* d_in, const int* in_sizes, int n_in,
                              void* d_out, int out_size, void* d_ws, size_t ws_size,
                              hipStream_t stream) {
  (void)in_sizes; (void)n_in; (void)out_size; (void)ws_size;
  const float* anchors = (const float*)d_in[0];
  const int*   labels  = (const int*)d_in[1];
  const float* proto   = (const float*)d_in[2];
  const int*   pmask   = (const int*)d_in[3];
  const float* localm  = (const float*)d_in[4];
  float* out = (float*)d_out;

  // Workspace layout: bf16 hi/lo copies (~44.6 MB) + f32/int scratch (~64 KB).
  char* w = (char*)d_ws;
  __bf16* a_hi = (__bf16*)w; w += (size_t)A_ * D_ * 2;
  __bf16* a_lo = (__bf16*)w; w += (size_t)A_ * D_ * 2;
  __bf16* p_hi = (__bf16*)w; w += (size_t)P_ * D_ * 2;
  __bf16* p_lo = (__bf16*)w; w += (size_t)P_ * D_ * 2;
  __bf16* l_hi = (__bf16*)w; w += (size_t)C_ * N_ * D_ * 2;
  __bf16* l_lo = (__bf16*)w; w += (size_t)C_ * N_ * D_ * 2;
  float* lmax    = (float*)w; w += (size_t)A_ * 4;
  float* negs    = (float*)w; w += (size_t)A_ * 4;
  float* mlpp    = (float*)w; w += (size_t)A_ * 4;
  int*   order   = (int*)w;   w += (size_t)A_ * 4;
  int*   cls_off = (int*)w;

  // One-pass f32 -> bf16 hi/lo splits (memory-bound, ~89 MB total traffic)
  kc_split<<<1024, 256, 0, stream>>>(anchors, a_hi, a_lo, A_ * D_);
  kc_split<<<256,  256, 0, stream>>>(proto,   p_hi, p_lo, P_ * D_);
  kc_split<<<2048, 256, 0, stream>>>(localm,  l_hi, l_lo, C_ * N_ * D_);

  k0_index<<<1, 32, 0, stream>>>(labels, order, cls_off);
  k1_protomax<<<A_ / (16 * 8), 256, 0, stream>>>(a_hi, a_lo, p_hi, p_lo, pmask, lmax);
  k2_neg<<<dim3(C_, 32), 256, 0, stream>>>(a_hi, a_lo, l_hi, l_lo, order, cls_off,
                                           lmax, negs);
  k3_pos<<<A_ / 8, 256, 0, stream>>>(anchors, proto, pmask, labels, lmax, negs, mlpp);
  k4_reduce<<<1, 256, 0, stream>>>(mlpp, out);
}
